// GPTGRNNDecoder_17506286698870
// MI455X (gfx1250) — compile-verified, matrix operand order checked by reading, not run on hardware
//
#include <hip/hip_runtime.h>
#include <math.h>

#define BN   64
#define SS   512
#define EE   512
#define HH   512
#define UU   256
#define MM   32
#define HEe  256
#define INP  543
#define INP_P 544

typedef __attribute__((ext_vector_type(16))) _Float16 v16h;
typedef __attribute__((ext_vector_type(8)))  _Float16 v8h;
typedef __attribute__((ext_vector_type(8)))  float    v8f;

// ---------------------------------------------------------------------------
// WMMA fragment load, ISA 7.12.2 16-bit A/B layout:
//  lane<16 : row = lane,    K chunks {k0+0..7, k0+16..23}
//  lane>=16: row = lane-16, K chunks {k0+8..15, k0+24..31}
// Two 16-byte loads per fragment (global_load_b128).
// ---------------------------------------------------------------------------
__device__ __forceinline__ v16h load_frag16(const _Float16* __restrict__ base,
                                            int ld, int row0, int k0, int lane) {
  int r  = row0 + (lane & 15);
  int kk = k0 + ((lane >> 4) << 3);
  const _Float16* p = base + (size_t)r * ld + kk;
  v8h lo = *reinterpret_cast<const v8h*>(p);
  v8h hi = *reinterpret_cast<const v8h*>(p + 16);
  v16h f;
#pragma unroll
  for (int t = 0; t < 8; ++t) { f[t] = lo[t]; f[t + 8] = hi[t]; }
  return f;
}

__device__ __forceinline__ float sigmoidf_(float x) { return 1.0f / (1.0f + expf(-x)); }

// ---------------------------------------------------------------------------
// Generic fused GRU layer (torch semantics), one time-step:
//   gi = x@wih^T + bih ; gh = h@whh^T + bhh
//   r = sig(gi_r+gh_r); z = sig(gi_z+gh_z); n = tanh(gi_n + r*gh_n)
//   h' = (1-z)*n + z*h
// 64 rows total; block = 128 threads = 4 waves, wave w -> rows 16w..16w+15,
// block computes one 16-column tile of the hidden state.
// ---------------------------------------------------------------------------
__global__ void __launch_bounds__(128) gru_layer_kernel(
    const _Float16* __restrict__ x, int ldx, int Kx,
    const _Float16* __restrict__ hp_h, const float* __restrict__ hp_f, int ldh,
    const _Float16* __restrict__ wih, const _Float16* __restrict__ whh,
    const float* __restrict__ bih, const float* __restrict__ bhh,
    float* __restrict__ hn_f, _Float16* __restrict__ hn_h, int ldn, int Hd)
{
  int lane = threadIdx.x & 31;
  int wave = threadIdx.x >> 5;
  int rowbase = wave * 16;
  int colbase = blockIdx.x * 16;
  int col = colbase + (lane & 15);

  float b_r = bih[col] + bhh[col];
  float b_z = bih[Hd + col] + bhh[Hd + col];
  float b_i = bih[2 * Hd + col];
  float b_h = bhh[2 * Hd + col];
  v8f acc_r, acc_z, acc_i, acc_h;
#pragma unroll
  for (int t = 0; t < 8; ++t) { acc_r[t] = b_r; acc_z[t] = b_z; acc_i[t] = b_i; acc_h[t] = b_h; }

  // input-hidden GEMM
  for (int k = 0; k < Kx; k += 32) {
    v16h a  = load_frag16(x,   ldx, rowbase,            k, lane);
    v16h br = load_frag16(wih, Kx,  colbase,            k, lane);
    v16h bz = load_frag16(wih, Kx,  Hd + colbase,       k, lane);
    v16h bn = load_frag16(wih, Kx,  2 * Hd + colbase,   k, lane);
    acc_r = __builtin_amdgcn_wmma_f32_16x16x32_f16(false, a, false, br, (short)0, acc_r, false, false);
    acc_z = __builtin_amdgcn_wmma_f32_16x16x32_f16(false, a, false, bz, (short)0, acc_z, false, false);
    acc_i = __builtin_amdgcn_wmma_f32_16x16x32_f16(false, a, false, bn, (short)0, acc_i, false, false);
  }
  // hidden-hidden GEMM
  for (int k = 0; k < Hd; k += 32) {
    v16h a  = load_frag16(hp_h, ldh, rowbase,          k, lane);
    v16h br = load_frag16(whh,  Hd,  colbase,          k, lane);
    v16h bz = load_frag16(whh,  Hd,  Hd + colbase,     k, lane);
    v16h bn = load_frag16(whh,  Hd,  2 * Hd + colbase, k, lane);
    acc_r = __builtin_amdgcn_wmma_f32_16x16x32_f16(false, a, false, br, (short)0, acc_r, false, false);
    acc_z = __builtin_amdgcn_wmma_f32_16x16x32_f16(false, a, false, bz, (short)0, acc_z, false, false);
    acc_h = __builtin_amdgcn_wmma_f32_16x16x32_f16(false, a, false, bn, (short)0, acc_h, false, false);
  }

  // C/D layout: lane holds column (lane&15), rows mb..mb+7
  int mb = rowbase + ((lane >> 4) << 3);
#pragma unroll
  for (int v = 0; v < 8; ++v) {
    float r  = sigmoidf_(acc_r[v]);
    float z  = sigmoidf_(acc_z[v]);
    float n  = tanhf(acc_i[v] + r * acc_h[v]);
    float hp = hp_f[(size_t)(mb + v) * ldh + col];
    float h  = (1.0f - z) * n + z * hp;
    hn_f[(size_t)(mb + v) * ldn + col] = h;
    hn_h[(size_t)(mb + v) * ldn + col] = (_Float16)h;
  }
}

// ---------------------------------------------------------------------------
// Generic linear: Y = act(X @ W^T + bias).  Rows must be multiple of 64.
// grid = (N/16, Mrows/64), block = 128 (4 waves, wave -> one 16-row tile).
// ---------------------------------------------------------------------------
__global__ void __launch_bounds__(128) linear_kernel(
    const _Float16* __restrict__ X, int ldx,
    const _Float16* __restrict__ W,
    const float* __restrict__ bias,
    float* __restrict__ Yf, _Float16* __restrict__ Yh, int ldy,
    int K, int act)
{
  int lane = threadIdx.x & 31;
  int wave = threadIdx.x >> 5;
  int rowbase = (blockIdx.y * 4 + wave) * 16;
  int colbase = blockIdx.x * 16;
  int col = colbase + (lane & 15);

  float b0 = bias ? bias[col] : 0.0f;
  v8f acc;
#pragma unroll
  for (int t = 0; t < 8; ++t) acc[t] = b0;

  for (int k = 0; k < K; k += 32) {
    v16h a = load_frag16(X, ldx, rowbase, k, lane);
    v16h b = load_frag16(W, K,   colbase, k, lane);
    acc = __builtin_amdgcn_wmma_f32_16x16x32_f16(false, a, false, b, (short)0, acc, false, false);
  }
  int mb = rowbase + ((lane >> 4) << 3);
#pragma unroll
  for (int v = 0; v < 8; ++v) {
    float y = acc[v];
    if (act == 1) y = fmaxf(y, 0.0f);
    if (Yf) Yf[(size_t)(mb + v) * ldy + col] = y;
    if (Yh) Yh[(size_t)(mb + v) * ldy + col] = (_Float16)y;
  }
}

// ---------------------------------------------------------------------------
// Attention scores: scores[b,s] = dot(tanh(W1e[b,s,:] + a[b,:] + cov[b,s]*wc), V)
// grid = (S/8, B), block 256 (8 waves, one wave per s).
// ---------------------------------------------------------------------------
__global__ void __launch_bounds__(256) score_kernel(
    const float* __restrict__ W1e, const float* __restrict__ aq,
    const float* __restrict__ cov, const float* __restrict__ wc,
    const float* __restrict__ V, float* __restrict__ scores)
{
  int lane = threadIdx.x & 31;
  int s = blockIdx.x * 8 + (threadIdx.x >> 5);
  int b = blockIdx.y;
  float c = cov[b * SS + s];
  const float* w1 = W1e + ((size_t)b * SS + s) * UU;
  const float* a  = aq + b * UU;
  float sum = 0.0f;
  for (int k = lane; k < UU; k += 32)
    sum += tanhf(w1[k] + a[k] + c * wc[k]) * V[k];
  for (int off = 16; off > 0; off >>= 1) sum += __shfl_xor(sum, off, 32);
  if (lane == 0) scores[b * SS + s] = sum;
}

// ---------------------------------------------------------------------------
// Masked softmax + coverage + hard gumbel (forward == one-hot argmax).
// One block per b; 256 threads handle s and s+256.
// ---------------------------------------------------------------------------
__global__ void __launch_bounds__(256) softmax_kernel(
    const float* __restrict__ scores, const int* __restrict__ lens,
    const float* __restrict__ gumbel, float* __restrict__ cov,
    float* __restrict__ covloss, float* __restrict__ pointer_out,
    int* __restrict__ sel, int step)
{
  __shared__ float red[256];
  __shared__ float rmax[256];
  __shared__ int   ridx[256];
  int b = blockIdx.x, t = threadIdx.x;
  int len = lens[b];
  float sc0 = (t < len)       ? scores[b * SS + t]       : -1e9f;
  float sc1 = (t + 256 < len) ? scores[b * SS + t + 256] : -1e9f;

  red[t] = fmaxf(sc0, sc1); __syncthreads();
  for (int off = 128; off > 0; off >>= 1) { if (t < off) red[t] = fmaxf(red[t], red[t + off]); __syncthreads(); }
  float m = red[0]; __syncthreads();

  float e0 = expf(sc0 - m), e1 = expf(sc1 - m);
  red[t] = e0 + e1; __syncthreads();
  for (int off = 128; off > 0; off >>= 1) { if (t < off) red[t] += red[t + off]; __syncthreads(); }
  float inv = 1.0f / red[0]; __syncthreads();
  float a0 = e0 * inv, a1 = e1 * inv;

  float c0 = cov[b * SS + t], c1 = cov[b * SS + t + 256];
  red[t] = fminf(a0, c0) + fminf(a1, c1); __syncthreads();
  for (int off = 128; off > 0; off >>= 1) { if (t < off) red[t] += red[t + off]; __syncthreads(); }
  if (t == 0) atomicAdd(covloss, red[0]);
  cov[b * SS + t]       = c0 + a0;
  cov[b * SS + t + 256] = c1 + a1;

  const float* g = gumbel + ((size_t)step * BN + b) * SS;
  float g0 = logf(a0 + 1e-12f) + g[t];
  float g1 = logf(a1 + 1e-12f) + g[t + 256];
  float gm; int gidx;
  if (g0 >= g1) { gm = g0; gidx = t; } else { gm = g1; gidx = t + 256; }
  rmax[t] = gm; ridx[t] = gidx; __syncthreads();
  for (int off = 128; off > 0; off >>= 1) {
    if (t < off && rmax[t + off] > rmax[t]) { rmax[t] = rmax[t + off]; ridx[t] = ridx[t + off]; }
    __syncthreads();
  }
  int amax = ridx[0];
  if (t == 0) sel[b] = amax;
  pointer_out[((size_t)b * SS + t)       * MM + step] = (t == amax)       ? 1.0f : 0.0f;
  pointer_out[((size_t)b * SS + t + 256) * MM + step] = (t + 256 == amax) ? 1.0f : 0.0f;
}

// fout layer 2: ein = sigmoid(hmid @ W2^T + b2); 64 rows x 128-dot, 1 block.
__global__ void __launch_bounds__(256) fout2_kernel(
    const float* __restrict__ hmid, const float* __restrict__ W2,
    const float* __restrict__ b2, _Float16* __restrict__ einx,
    float* __restrict__ theta, int j)
{
  int t = threadIdx.x;
  int row = t >> 2, q = t & 3;
  float s = 0.0f;
  for (int k = q * 32; k < q * 32 + 32; ++k) s += hmid[row * 128 + k] * W2[k];
  s += __shfl_xor(s, 1, 32);
  s += __shfl_xor(s, 2, 32);
  if (q == 0) {
    float v = sigmoidf_(s + b2[0]);
    einx[row * 32]      = (_Float16)v;   // column 0 of next edge-GRU input
    theta[row * 31 + j] = v;
  }
}

// write adj column `step` and build next dec_in = [chosen(enc row sel), theta, 0pad]
__global__ void __launch_bounds__(256) adj_decin_kernel(
    const float* __restrict__ theta, const int* __restrict__ sel,
    const _Float16* __restrict__ enc16, float* __restrict__ adj_out,
    _Float16* __restrict__ dec_in, int step)
{
  int b = blockIdx.x, t = threadIdx.x;
  if (t < 31) adj_out[(size_t)b * 31 * 32 + t * 32 + step] = theta[b * 31 + t];
  int sb = sel[b];
  for (int k = t; k < INP_P; k += 256) {
    _Float16 v;
    if (k < EE)        v = enc16[((size_t)b * SS + sb) * EE + k];
    else if (k < INP)  v = (_Float16)theta[b * 31 + (k - EE)];
    else               v = (_Float16)0.0f;
    dec_in[b * INP_P + k] = v;
  }
}

__global__ void gather_enc_kernel(const float* __restrict__ h,
    const int* __restrict__ lens, const int* __restrict__ offs,
    float* __restrict__ enc_f, _Float16* __restrict__ enc_h)
{
  size_t idx = (size_t)blockIdx.x * 256 + threadIdx.x;
  int e = (int)(idx % EE);
  size_t rs = idx / EE;
  int s = (int)(rs % SS);
  int b = (int)(rs / SS);
  float v = 0.0f;
  if (s < lens[b]) v = h[((size_t)offs[b] + s) * EE + e];
  enc_f[idx] = v;
  enc_h[idx] = (_Float16)v;
}

__global__ void prefix_kernel(const int* __restrict__ lens, int* __restrict__ offs) {
  if (threadIdx.x == 0) { int a = 0; for (int b = 0; b < BN; ++b) { offs[b] = a; a += lens[b]; } }
}

__global__ void cvt_kernel(const float* __restrict__ s, _Float16* __restrict__ d, int n) {
  int i = blockIdx.x * 256 + threadIdx.x;
  if (i < n) d[i] = (_Float16)s[i];
}
__global__ void cvt_pad_kernel(const float* __restrict__ s, _Float16* __restrict__ d,
                               int rows, int kin, int kout) {
  int i = blockIdx.x * 256 + threadIdx.x;
  if (i < rows * kout) {
    int r = i / kout, k = i % kout;
    d[i] = (k < kin) ? (_Float16)s[r * kin + k] : (_Float16)0.0f;
  }
}
__global__ void zero_f_kernel(float* p, int n) { int i = blockIdx.x * 256 + threadIdx.x; if (i < n) p[i] = 0.0f; }
__global__ void zero_h_kernel(_Float16* p, int n) { int i = blockIdx.x * 256 + threadIdx.x; if (i < n) p[i] = (_Float16)0.0f; }
__global__ void finalize_kernel(const float* cl, float* out) { if (threadIdx.x == 0) out[0] = cl[0] / (float)BN; }

// ---------------------------------------------------------------------------
extern "C" void kernel_launch(void* const* d_in, const int* in_sizes, int n_in,
                              void* d_out, int out_size, void* d_ws, size_t ws_size,
                              hipStream_t stream) {
  (void)in_sizes; (void)n_in; (void)out_size; (void)ws_size;
  const int*   lens     = (const int*)d_in[0];
  const float* h_in     = (const float*)d_in[1];
  const float* hg       = (const float*)d_in[2];
  const float* gumbel   = (const float*)d_in[3];
  const float* proj_W   = (const float*)d_in[4];
  const float* proj_b   = (const float*)d_in[5];
  const float* g0_wih   = (const float*)d_in[6];
  const float* g0_whh   = (const float*)d_in[7];
  const float* g0_bih   = (const float*)d_in[8];
  const float* g0_bhh   = (const float*)d_in[9];
  const float* g1_wih   = (const float*)d_in[10];
  const float* g1_whh   = (const float*)d_in[11];
  const float* g1_bih   = (const float*)d_in[12];
  const float* g1_bhh   = (const float*)d_in[13];
  const float* att_W1   = (const float*)d_in[14];
  const float* att_W2   = (const float*)d_in[15];
  const float* att_wc   = (const float*)d_in[16];
  const float* att_V    = (const float*)d_in[17];
  const float* eproj_W  = (const float*)d_in[18];
  const float* eproj_b  = (const float*)d_in[19];
  const float* e0_wih   = (const float*)d_in[20];
  const float* e0_whh   = (const float*)d_in[21];
  const float* e0_bih   = (const float*)d_in[22];
  const float* e0_bhh   = (const float*)d_in[23];
  const float* e1_wih   = (const float*)d_in[24];
  const float* e1_whh   = (const float*)d_in[25];
  const float* e1_bih   = (const float*)d_in[26];
  const float* e1_bhh   = (const float*)d_in[27];
  const float* fo_W1    = (const float*)d_in[28];
  const float* fo_b1    = (const float*)d_in[29];
  const float* fo_W2    = (const float*)d_in[30];
  const float* fo_b2    = (const float*)d_in[31];

  float* out_ptr     = (float*)d_out;                          // pointer [64,512,32]
  float* out_covloss = out_ptr + (size_t)BN * SS * MM;         // scalar
  float* out_enc     = out_covloss + 1;                        // encoder_out [64,512,512]
  float* out_adj     = out_enc + (size_t)BN * SS * EE;         // adj [64,31,32]

  // ---- workspace carve ----
  char* wp = (char*)d_ws;
  auto alloc = [&](size_t bytes) -> char* {
    char* p = wp; wp += (bytes + 255) & ~(size_t)255; return p;
  };
  _Float16* enc16   = (_Float16*)alloc((size_t)BN * SS * EE * 2);
  float*    W1e     = (float*)   alloc((size_t)BN * SS * UU * 4);
  _Float16* w_proj  = (_Float16*)alloc((size_t)1024 * 512 * 2);
  _Float16* w_g0ih  = (_Float16*)alloc((size_t)1536 * INP_P * 2);
  _Float16* w_g0hh  = (_Float16*)alloc((size_t)1536 * 512 * 2);
  _Float16* w_g1ih  = (_Float16*)alloc((size_t)1536 * 512 * 2);
  _Float16* w_g1hh  = (_Float16*)alloc((size_t)1536 * 512 * 2);
  _Float16* w_att1  = (_Float16*)alloc((size_t)256 * 512 * 2);
  _Float16* w_att2  = (_Float16*)alloc((size_t)256 * 512 * 2);
  _Float16* w_eproj = (_Float16*)alloc((size_t)512 * 1024 * 2);
  _Float16* w_e0ih  = (_Float16*)alloc((size_t)768 * 32 * 2);
  _Float16* w_e0hh  = (_Float16*)alloc((size_t)768 * 256 * 2);
  _Float16* w_e1ih  = (_Float16*)alloc((size_t)768 * 256 * 2);
  _Float16* w_e1hh  = (_Float16*)alloc((size_t)768 * 256 * 2);
  _Float16* w_fo1   = (_Float16*)alloc((size_t)128 * 256 * 2);
  _Float16* hg16    = (_Float16*)alloc((size_t)BN * 512 * 2);
  _Float16* dec_in16= (_Float16*)alloc((size_t)BN * INP_P * 2);
  float*    dhA_f   = (float*)   alloc((size_t)BN * 1024 * 4);
  _Float16* dhA_h   = (_Float16*)alloc((size_t)BN * 1024 * 2);
  float*    dhB_f   = (float*)   alloc((size_t)BN * 1024 * 4);
  _Float16* dhB_h   = (_Float16*)alloc((size_t)BN * 1024 * 2);
  float*    a_q     = (float*)   alloc((size_t)BN * UU * 4);
  float*    cov     = (float*)   alloc((size_t)BN * SS * 4);
  float*    scores  = (float*)   alloc((size_t)BN * SS * 4);
  int*      sel     = (int*)     alloc((size_t)BN * 4);
  float*    covloss = (float*)   alloc(4);
  int*      offs    = (int*)     alloc((size_t)BN * 4);
  float*    ehA_f   = (float*)   alloc((size_t)BN * 512 * 4);
  _Float16* ehA_h   = (_Float16*)alloc((size_t)BN * 512 * 2);
  float*    ehB_f   = (float*)   alloc((size_t)BN * 512 * 4);
  _Float16* ehB_h   = (_Float16*)alloc((size_t)BN * 512 * 2);
  _Float16* einx    = (_Float16*)alloc((size_t)BN * 32 * 2);
  float*    hmid    = (float*)   alloc((size_t)BN * 128 * 4);
  float*    theta   = (float*)   alloc((size_t)BN * 31 * 4);

  auto cgrid = [](int n) { return dim3((n + 255) / 256); };

  // ---- one-time per call: conversions / init ----
  prefix_kernel<<<1, 64, 0, stream>>>(lens, offs);
  cvt_kernel<<<cgrid(1024*512), 256, 0, stream>>>(proj_W, w_proj, 1024*512);
  cvt_pad_kernel<<<cgrid(1536*INP_P), 256, 0, stream>>>(g0_wih, w_g0ih, 1536, INP, INP_P);
  cvt_kernel<<<cgrid(1536*512), 256, 0, stream>>>(g0_whh, w_g0hh, 1536*512);
  cvt_kernel<<<cgrid(1536*512), 256, 0, stream>>>(g1_wih, w_g1ih, 1536*512);
  cvt_kernel<<<cgrid(1536*512), 256, 0, stream>>>(g1_whh, w_g1hh, 1536*512);
  cvt_kernel<<<cgrid(256*512), 256, 0, stream>>>(att_W1, w_att1, 256*512);
  cvt_kernel<<<cgrid(256*512), 256, 0, stream>>>(att_W2, w_att2, 256*512);
  cvt_kernel<<<cgrid(512*1024), 256, 0, stream>>>(eproj_W, w_eproj, 512*1024);
  cvt_pad_kernel<<<cgrid(768*32), 256, 0, stream>>>(e0_wih, w_e0ih, 768, 1, 32);
  cvt_kernel<<<cgrid(768*256), 256, 0, stream>>>(e0_whh, w_e0hh, 768*256);
  cvt_kernel<<<cgrid(768*256), 256, 0, stream>>>(e1_wih, w_e1ih, 768*256);
  cvt_kernel<<<cgrid(768*256), 256, 0, stream>>>(e1_whh, w_e1hh, 768*256);
  cvt_kernel<<<cgrid(128*256), 256, 0, stream>>>(fo_W1, w_fo1, 128*256);
  cvt_kernel<<<cgrid(BN*512), 256, 0, stream>>>(hg, hg16, BN*512);

  zero_f_kernel<<<cgrid(BN*SS), 256, 0, stream>>>(cov, BN*SS);
  zero_f_kernel<<<1, 256, 0, stream>>>(covloss, 1);
  zero_h_kernel<<<cgrid(BN*INP_P), 256, 0, stream>>>(dec_in16, BN*INP_P);

  // encoder_out: gather/pad (f32 to output, f16 mirror for WMMA)
  gather_enc_kernel<<<dim3((BN*SS*EE)/256), 256, 0, stream>>>(h_in, lens, offs, out_enc, enc16);

  // dec_h init: relu(hg @ proj_W^T + b), layer l -> columns [l*H, (l+1)*H) of dhA
  for (int l = 0; l < 2; ++l)
    linear_kernel<<<dim3(HH/16, 1), 128, 0, stream>>>(hg16, 512, w_proj + (size_t)l*HH*512,
        proj_b + l*HH, dhA_f + l*HH, dhA_h + l*HH, 1024, 512, 1);

  // W1e = encoder_out @ att_W1^T  (32768 x 256 x 512)
  linear_kernel<<<dim3(UU/16, (BN*SS)/64), 128, 0, stream>>>(enc16, EE, w_att1,
      (const float*)nullptr, W1e, (_Float16*)nullptr, UU, 512, 0);

  float*    dhp_f = dhA_f; _Float16* dhp_h = dhA_h;
  float*    dhn_f = dhB_f; _Float16* dhn_h = dhB_h;

  for (int i = 0; i < MM; ++i) {
    // graph GRU, layer 0 then layer 1 (double-buffered)
    gru_layer_kernel<<<dim3(HH/16), 128, 0, stream>>>(dec_in16, INP_P, INP_P,
        dhp_h, dhp_f, 1024, w_g0ih, w_g0hh, g0_bih, g0_bhh, dhn_f, dhn_h, 1024, HH);
    gru_layer_kernel<<<dim3(HH/16), 128, 0, stream>>>(dhn_h, 1024, 512,
        dhp_h + HH, dhp_f + HH, 1024, w_g1ih, w_g1hh, g1_bih, g1_bhh,
        dhn_f + HH, dhn_h + HH, 1024, HH);

    // attention
    linear_kernel<<<dim3(UU/16, 1), 128, 0, stream>>>(dhn_h + HH, 1024, w_att2,
        (const float*)nullptr, a_q, (_Float16*)nullptr, UU, 512, 0);
    score_kernel<<<dim3(SS/8, BN), 256, 0, stream>>>(W1e, a_q, cov, att_wc, att_V, scores);
    softmax_kernel<<<dim3(BN), 256, 0, stream>>>(scores, lens, gumbel, cov, covloss,
        out_ptr, sel, i);

    // edge RNN init
    zero_f_kernel<<<cgrid(BN*31), 256, 0, stream>>>(theta, BN*31);
    zero_h_kernel<<<cgrid(BN*32), 256, 0, stream>>>(einx, BN*32);
    for (int l = 0; l < 2; ++l)
      linear_kernel<<<dim3(HEe/16, 1), 128, 0, stream>>>(dhn_h, 1024,
          w_eproj + (size_t)l*HEe*1024, eproj_b + l*HEe,
          ehA_f + l*HEe, ehA_h + l*HEe, 512, 1024, 1);

    float*    ehp_f = ehA_f; _Float16* ehp_h = ehA_h;
    float*    ehn_f = ehB_f; _Float16* ehn_h = ehB_h;
    for (int j = 0; j < i; ++j) {
      gru_layer_kernel<<<dim3(HEe/16), 128, 0, stream>>>(einx, 32, 32,
          ehp_h, ehp_f, 512, w_e0ih, w_e0hh, e0_bih, e0_bhh, ehn_f, ehn_h, 512, HEe);
      gru_layer_kernel<<<dim3(HEe/16), 128, 0, stream>>>(ehn_h, 512, 256,
          ehp_h + HEe, ehp_f + HEe, 512, w_e1ih, w_e1hh, e1_bih, e1_bhh,
          ehn_f + HEe, ehn_h + HEe, 512, HEe);
      linear_kernel<<<dim3(128/16, 1), 128, 0, stream>>>(ehn_h + HEe, 512, w_fo1,
          fo_b1, hmid, (_Float16*)nullptr, 128, 256, 1);
      fout2_kernel<<<1, 256, 0, stream>>>(hmid, fo_W2, fo_b2, einx, theta, j);
      { float* tf = ehp_f; ehp_f = ehn_f; ehn_f = tf; }
      { _Float16* th = ehp_h; ehp_h = ehn_h; ehn_h = th; }
    }

    // adj column i + next dec_in (chosen row gather + theta + zero pad)
    adj_decin_kernel<<<dim3(BN), 256, 0, stream>>>(theta, sel, enc16, out_adj, dec_in16, i);

    { float* tf = dhp_f; dhp_f = dhn_f; dhn_f = tf; }
    { _Float16* th = dhp_h; dhp_h = dhn_h; dhn_h = th; }
  }

  finalize_kernel<<<1, 64, 0, stream>>>(covloss, out_covloss);
}
